// VAEBase_79637283602456
// MI455X (gfx1250) — compile-verified, matrix operand order checked by reading, not run on hardware
//
#include <hip/hip_runtime.h>
#include <hip/hip_bf16.h>

typedef __attribute__((ext_vector_type(16))) _Float16     v16h;
typedef __attribute__((ext_vector_type(8)))  float        v8f;
typedef __attribute__((ext_vector_type(8)))  unsigned int v8u;

static constexpr int Bsz = 8, Ssz = 1024, Vsz = 50257, Hsz = 1024, Lsz = 128;
static constexpr int Rows = Bsz * Ssz;  // 8192

// ---------------------------------------------------------------------------
// helpers
// ---------------------------------------------------------------------------
__device__ inline unsigned int fsort(float x) {
  unsigned int u = __float_as_uint(x);
  return (u & 0x80000000u) ? ~u : (u | 0x80000000u);
}

// Load a 16-bit WMMA fragment (A or B) for one lane from a row-major row in
// global memory, viewed as u32 K-pairs. Per the ISA 16-bit 16x32 layout,
// VGPR g holds the K-pair at pair-index  k0/2 + g + (g&4) + 4*(lane>=16).
__device__ inline v16h load_frag16(const unsigned int* rowPtr, int k0half, int half4) {
  v8u u;
#pragma unroll
  for (int g = 0; g < 8; ++g) u[g] = rowPtr[k0half + g + (g & 4) + half4];
  return __builtin_bit_cast(v16h, u);
}

// ---------------------------------------------------------------------------
// 1) normalize embedding rows: e = emb/max(||emb||,eps)*sqrt(H)
// ---------------------------------------------------------------------------
__global__ __launch_bounds__(256) void prep_emb(const float* __restrict__ emb,
                                                float* __restrict__ e32,
                                                _Float16* __restrict__ e16,
                                                float* __restrict__ halfnorm) {
  int v = blockIdx.x;
  const float* row = emb + (size_t)v * Hsz;
  float vals[4];
  float ss = 0.f;
#pragma unroll
  for (int i = 0; i < 4; ++i) {
    float t = row[threadIdx.x + i * 256];
    vals[i] = t;
    ss += t * t;
  }
  __shared__ float red[256];
  red[threadIdx.x] = ss;
  __syncthreads();
  for (int s = 128; s > 0; s >>= 1) {
    if (threadIdx.x < s) red[threadIdx.x] += red[threadIdx.x + s];
    __syncthreads();
  }
  float tot = red[0];
  float scale = 32.0f / fmaxf(sqrtf(tot), 1e-12f);  // sqrt(1024)=32
#pragma unroll
  for (int i = 0; i < 4; ++i) {
    int c = threadIdx.x + i * 256;
    float ev = vals[i] * scale;
    e32[(size_t)v * Hsz + c] = ev;
    e16[(size_t)v * Hsz + c] = (_Float16)ev;
  }
  if (threadIdx.x == 0) halfnorm[v] = 0.5f * tot * scale * scale;
}

// ---------------------------------------------------------------------------
// 2) elementwise f32 -> f16 cast
// ---------------------------------------------------------------------------
__global__ __launch_bounds__(256) void cast_f32_f16(const float* __restrict__ in,
                                                    _Float16* __restrict__ out, int n) {
  int i = blockIdx.x * 1024 + threadIdx.x;
#pragma unroll
  for (int j = 0; j < 4; ++j) {
    int k = i + j * 256;
    if (k < n) out[k] = (_Float16)in[k];
  }
}

// ---------------------------------------------------------------------------
// 3) WMMA GEMM: C[r][c] = act( dot(A_row r, B_row c) + bias[c] ), f16 out.
// ---------------------------------------------------------------------------
template <int KDIM, bool GELU>
__global__ __launch_bounds__(256) void gemm_bias(const _Float16* __restrict__ A,
                                                 const _Float16* __restrict__ B,
                                                 const float* __restrict__ bias,
                                                 _Float16* __restrict__ C, int N) {
  int wave = threadIdx.x >> 5;
  int lane = threadIdx.x & 31;
  int half = lane >> 4, l15 = lane & 15, half4 = half * 4;
  int ntiles = N >> 4;
  int t = blockIdx.x * 8 + wave;
  int mt = t / ntiles, nt = t % ntiles;

  const unsigned int* Arow = (const unsigned int*)(A + (size_t)(mt * 16 + l15) * KDIM);
  const unsigned int* Brow = (const unsigned int*)(B + (size_t)(nt * 16 + l15) * KDIM);

  v8f acc = {0.f, 0.f, 0.f, 0.f, 0.f, 0.f, 0.f, 0.f};
#pragma unroll 4
  for (int k0 = 0; k0 < KDIM; k0 += 32) {
    v16h a = load_frag16(Arow, k0 >> 1, half4);
    v16h b = load_frag16(Brow, k0 >> 1, half4);
    acc = __builtin_amdgcn_wmma_f32_16x16x32_f16(false, a, false, b, (short)0, acc,
                                                 false, false);
  }
  int col = nt * 16 + l15;
  float bv = bias[col];
#pragma unroll
  for (int j = 0; j < 8; ++j) {
    int r = mt * 16 + j + half * 8;  // C/D layout: VGPR j -> M=j (+8 upper half)
    float c = acc[j] + bv;
    if (GELU) c = 0.5f * c * (1.0f + erff(c * 0.70710678118654752f));
    C[(size_t)r * N + col] = (_Float16)c;
  }
}

// ---------------------------------------------------------------------------
// 4) zero the packed argmax buffer
// ---------------------------------------------------------------------------
__global__ void init_nn(unsigned long long* nnpack) {
  nnpack[blockIdx.x * 256 + threadIdx.x] = 0ull;
}

// ---------------------------------------------------------------------------
// 5) scoring GEMM + argmax, LDS-staged via async copies.
//    score[r][v] = p_r . e_v - 0.5*||e_v||^2 ; argmin(dist)==argmax(score).
//    Block: 8 waves x 16 rows = 128 rows; group = 64 v columns.
//    Per phase (128 K): e-tile (64 rows x 128 K f16, 80B padded row stride)
//    is copied to LDS with global_load_async_to_lds_b128 (ASYNCcnt), double
//    buffered so the copy of phase p+1 overlaps the 16 WMMAs of phase p.
// ---------------------------------------------------------------------------
__global__ __launch_bounds__(256) void score_argmax(const _Float16* __restrict__ P,
                                                    const _Float16* __restrict__ E,
                                                    const float* __restrict__ halfnorm,
                                                    unsigned long long* __restrict__ nnpack,
                                                    int groupsPerSplit) {
  // 2 buffers x 4 chunks x 64 rows x 80 bytes = 40960 bytes
  __shared__ alignas(16) unsigned char stage[2 * 20480];

  const int tid = threadIdx.x;
  int wave = tid >> 5, lane = tid & 31;
  int half = lane >> 4, l15 = lane & 15, half4 = half * 4;
  int row0 = blockIdx.x * 128 + wave * 16;
  const unsigned int* Arow = (const unsigned int*)(P + (size_t)(row0 + l15) * Hsz);

  int totalGroups = (Vsz + 63) >> 6;
  int gbeg = blockIdx.y * groupsPerSplit;
  int gend = min(gbeg + groupsPerSplit, totalGroups);

  // issue this thread's 4 async b128 copies for one phase of one group
  auto stage_phase = [&](int vbase, int kp, int buf) {
#pragma unroll
    for (int i = 0; i < 4; ++i) {
      int q = tid + i * 256;           // 1024 pieces of 16B per phase
      int c = q >> 8;                  // k-chunk 0..3
      int r = (q >> 2) & 63;           // v-row in group
      int s = q & 3;                   // 16B piece within the 64B row payload
      int vrow = min(vbase + r, Vsz - 1);  // clamp OOB rows (masked later)
      unsigned long long g64 =
          (unsigned long long)((const unsigned char*)E +
                               ((size_t)vrow * Hsz + (size_t)(kp + c * 32)) * 2 +
                               (size_t)(s * 16));
      unsigned ldst = (unsigned)(size_t)(stage + buf * 20480 + c * 5120 + r * 80 + s * 16);
      asm volatile("global_load_async_to_lds_b128 %0, %1, off" ::"v"(ldst), "v"(g64)
                   : "memory");
    }
  };

  unsigned long long best[8];
#pragma unroll
  for (int j = 0; j < 8; ++j) best[j] = 0ull;  // sortable-min sentinel

  if (gbeg < gend) stage_phase(gbeg << 6, 0, 0);  // prologue copy

  for (int g = gbeg; g < gend; ++g) {
    int vbase = g << 6;
    v8f acc[4];
#pragma unroll
    for (int t = 0; t < 4; ++t) acc[t] = (v8f){0.f, 0.f, 0.f, 0.f, 0.f, 0.f, 0.f, 0.f};

    for (int ph = 0; ph < 8; ++ph) {  // 8 phases x 128 K = 1024 K
      int buf = ph & 1;
      asm volatile("s_wait_asynccnt 0" ::: "memory");  // my copies for buf done
      __syncthreads();                                 // everyone's copies + reads

      // issue next phase (or next group's phase 0) into the other buffer;
      // that buffer was last read in the previous phase, finished by all waves at the barrier -> safe.
      if (ph + 1 < 8)
        stage_phase(vbase, (ph + 1) * 128, buf ^ 1);
      else if (g + 1 < gend)
        stage_phase((g + 1) << 6, 0, buf ^ 1);

      // hoist the 4 A fragments of this phase; loads overlap LDS reads
      v16h a[4];
#pragma unroll
      for (int c = 0; c < 4; ++c) a[c] = load_frag16(Arow, ph * 64 + c * 16, half4);

      const unsigned char* bufp = stage + buf * 20480;
#pragma unroll
      for (int c = 0; c < 4; ++c) {
#pragma unroll
        for (int t = 0; t < 4; ++t) {
          const unsigned char* rp = bufp + c * 5120 + (t * 16 + l15) * 80 + half * 16;
          uint4 lo = *(const uint4*)rp;         // ds_load_b128, conflict-free
          uint4 hi = *(const uint4*)(rp + 32);  // (lanes spaced 20 banks)
          v8u u;
          u[0] = lo.x; u[1] = lo.y; u[2] = lo.z; u[3] = lo.w;
          u[4] = hi.x; u[5] = hi.y; u[6] = hi.z; u[7] = hi.w;
          v16h b = __builtin_bit_cast(v16h, u);
          acc[t] = __builtin_amdgcn_wmma_f32_16x16x32_f16(false, a[c], false, b,
                                                          (short)0, acc[t], false,
                                                          false);
        }
      }
    }

    // epilogue: fold -0.5*||e||^2, track packed (score, ~v) max per row
#pragma unroll
    for (int t = 0; t < 4; ++t) {
      int vcol = vbase + t * 16 + l15;
      if (vcol < Vsz) {
        float hn = halfnorm[vcol];
        unsigned int vinv = 0xFFFFFFFFu - (unsigned int)vcol;  // tie -> lowest v
#pragma unroll
        for (int j = 0; j < 8; ++j) {
          float s = acc[t][j] - hn;
          unsigned long long pk = ((unsigned long long)fsort(s) << 32) | vinv;
          if (pk > best[j]) best[j] = pk;
        }
      }
    }
  }

  // reduce across the 16 lanes that share each output row, then global merge
#pragma unroll
  for (int j = 0; j < 8; ++j) {
    unsigned long long b = best[j];
    for (int off = 8; off >= 1; off >>= 1) {
      unsigned long long o = __shfl_xor(b, off, 16);
      if (o > b) b = o;
    }
    if (l15 == 0) atomicMax(&nnpack[row0 + j + half * 8], b);
  }
}

// ---------------------------------------------------------------------------
// 6) gather winning fp32 e-rows into the output
// ---------------------------------------------------------------------------
__global__ __launch_bounds__(256) void gather_rows(const unsigned long long* __restrict__ nnpack,
                                                   const float* __restrict__ e32,
                                                   float* __restrict__ out) {
  int r = blockIdx.x;
  unsigned int idx = 0xFFFFFFFFu - (unsigned int)(nnpack[r] & 0xFFFFFFFFull);
  if (idx >= (unsigned int)Vsz) idx = 0;
  const float4* src = (const float4*)(e32 + (size_t)idx * Hsz);
  float4* dst = (float4*)(out + (size_t)r * Hsz);
  dst[threadIdx.x] = src[threadIdx.x];
}

// ---------------------------------------------------------------------------
extern "C" void kernel_launch(void* const* d_in, const int* in_sizes, int n_in,
                              void* d_out, int out_size, void* d_ws, size_t ws_size,
                              hipStream_t stream) {
  const float* x   = (const float*)d_in[0];
  // d_in[1] = padding_mask (unused by the reference computation)
  const float* emb = (const float*)d_in[2];
  const float* w1  = (const float*)d_in[3];
  const float* b1  = (const float*)d_in[4];
  const float* w2  = (const float*)d_in[5];
  const float* b2  = (const float*)d_in[6];

  char* ws = (char*)d_ws;
  size_t off = 0;
  auto alloc = [&](size_t bytes) -> char* {
    char* p = ws + off;
    off = (off + bytes + 255) & ~(size_t)255;
    return p;
  };
  float*              e32      = (float*)alloc((size_t)Vsz * Hsz * 4);
  _Float16*           e16      = (_Float16*)alloc((size_t)Vsz * Hsz * 2);
  float*              halfnorm = (float*)alloc((size_t)Vsz * 4);
  _Float16*           x16      = (_Float16*)alloc((size_t)Rows * Lsz * 2);
  _Float16*           w116     = (_Float16*)alloc((size_t)Hsz * Lsz * 2);
  _Float16*           w216     = (_Float16*)alloc((size_t)Hsz * Hsz * 2);
  _Float16*           h16      = (_Float16*)alloc((size_t)Rows * Hsz * 2);
  _Float16*           p16      = (_Float16*)alloc((size_t)Rows * Hsz * 2);
  unsigned long long* nnpack   = (unsigned long long*)alloc((size_t)Rows * 8);

  prep_emb<<<Vsz, 256, 0, stream>>>(emb, e32, e16, halfnorm);

  cast_f32_f16<<<(Rows * Lsz + 1023) / 1024, 256, 0, stream>>>(x, x16, Rows * Lsz);
  cast_f32_f16<<<(Hsz * Lsz + 1023) / 1024, 256, 0, stream>>>(w1, w116, Hsz * Lsz);
  cast_f32_f16<<<(Hsz * Hsz + 1023) / 1024, 256, 0, stream>>>(w2, w216, Hsz * Hsz);

  // h = gelu(x @ w1^T + b1)   [8192x1024, K=128]
  gemm_bias<Lsz, true><<<(Rows / 16) * (Hsz / 16) / 8, 256, 0, stream>>>(x16, w116, b1,
                                                                        h16, Hsz);
  // p = h @ w2^T + b2         [8192x1024, K=1024]
  gemm_bias<Hsz, false><<<(Rows / 16) * (Hsz / 16) / 8, 256, 0, stream>>>(h16, w216, b2,
                                                                         p16, Hsz);

  init_nn<<<Rows / 256, 256, 0, stream>>>(nnpack);

  const int NSPLIT = 10;
  int totalGroups = (Vsz + 63) / 64;              // 786 groups of 64 columns
  int gps = (totalGroups + NSPLIT - 1) / NSPLIT;  // groups per v-split
  dim3 sgrid(Rows / 128, NSPLIT);
  score_argmax<<<sgrid, 256, 0, stream>>>(p16, e16, halfnorm, nnpack, gps);

  gather_rows<<<Rows, 256, 0, stream>>>(nnpack, e32, (float*)d_out);
}
